// ManLSTM_38826504356368
// MI455X (gfx1250) — compile-verified, matrix operand order checked by reading, not run on hardware
//
#include <hip/hip_runtime.h>

// Problem dims (fixed by the reference)
#define B_  64
#define T_  512
#define I_  256
#define H_  512

typedef float  v8f   __attribute__((ext_vector_type(8)));
typedef __bf16 v16bf __attribute__((ext_vector_type(16)));
typedef unsigned v4u __attribute__((ext_vector_type(4)));
// Exact pointee type expected by the async-LDS builtin (per hipcc diagnostic):
typedef int v4i_raw __attribute__((__vector_size__(4 * sizeof(int))));

#define AS_GLOBAL __attribute__((address_space(1)))
#define AS_LDS    __attribute__((address_space(3)))

// ---------------------------------------------------------------------------
// fp32 -> bf16 bits, round-to-nearest-even (raw u16 storage everywhere so no
// scalar __bf16 arithmetic support is needed).
// ---------------------------------------------------------------------------
__device__ __forceinline__ unsigned short f2bf(float f) {
    union { float f; unsigned u; } v; v.f = f;
    unsigned u = v.u;
    unsigned r = u + 0x7FFFu + ((u >> 16) & 1u);
    return (unsigned short)(r >> 16);
}

// ---------------------------------------------------------------------------
// 16B global -> LDS copy via the CDNA5 async-DMA path
// (global_load_async_to_lds_b128, tracked by ASYNCcnt); plain load + ds_store
// fallback if the builtin is not declared.
// ---------------------------------------------------------------------------
__device__ __forceinline__ void copy16_g2l(const unsigned short* g, unsigned short* l) {
#if __has_builtin(__builtin_amdgcn_global_load_async_to_lds_b128)
    __builtin_amdgcn_global_load_async_to_lds_b128(
        (AS_GLOBAL v4i_raw*)(uintptr_t)g,
        (AS_LDS v4i_raw*)(unsigned)(uintptr_t)l, 0, 0);
#else
    *(v4u*)l = *(const v4u*)g;
#endif
}

__device__ __forceinline__ void wait_g2l() {
#if __has_builtin(__builtin_amdgcn_global_load_async_to_lds_b128)
#if __has_builtin(__builtin_amdgcn_s_wait_asynccnt)
    __builtin_amdgcn_s_wait_asynccnt(0);
#else
    asm volatile("s_wait_asynccnt 0x0" ::: "memory");
#endif
#endif
    // fallback path: __syncthreads() in the caller orders the ds_stores
}

// ---------------------------------------------------------------------------
// Pack weights into WMMA B-matrix tile layout (bf16).
//   dst index = (((s*4 + g)*KT + kt)*32 + lane)*16 + q
//   lane<16 : N = 16*s + lane,    K = 32*kt + (q<8 ? q : q+8)
//   lane>=16: N = 16*s + lane-16, K = 32*kt + (q<8 ? q : q+8) + 8
//   K < KX rows come from Wx (4,KX,H), rest from Wh (4,H,H). Gate order i,f,o,g.
// ---------------------------------------------------------------------------
__global__ void pack_w_kernel(const float* __restrict__ Wx,
                              const float* __restrict__ Wh,
                              unsigned short* __restrict__ dst,
                              int KX, int KT) {
    long idx = (long)blockIdx.x * blockDim.x + threadIdx.x;
    long total = 32L * 4 * KT * 32 * 16;
    if (idx >= total) return;
    int q    = (int)(idx & 15);
    int lane = (int)((idx >> 4) & 31);
    long hi  = idx >> 9;                 // (s*4+g)*KT + kt
    int kt   = (int)(hi % KT);
    long sg  = hi / KT;
    int g    = (int)(sg & 3);
    int s    = (int)(sg >> 2);
    int n    = s * 16 + (lane & 15);
    int k    = kt * 32 + ((q < 8) ? q : q + 8) + ((lane >= 16) ? 8 : 0);
    float val = (k < KX) ? Wx[((long)g * KX + k) * H_ + n]
                         : Wh[((long)g * H_ + (k - KX)) * H_ + n];
    dst[idx] = f2bf(val);
}

// ---------------------------------------------------------------------------
// Pack x (B,T,I) fp32 into per-timestep WMMA A-matrix tile layout (bf16).
//   Xp index = (((t*4 + mt)*8 + kt)*32 + lane)*16 + q
// ---------------------------------------------------------------------------
__global__ void pack_x_kernel(const float* __restrict__ x,
                              unsigned short* __restrict__ Xp) {
    long idx = (long)blockIdx.x * blockDim.x + threadIdx.x;
    const long total = (long)T_ * 4 * 8 * 32 * 16;
    if (idx >= total) return;
    int q    = (int)(idx & 15);
    int lane = (int)((idx >> 4) & 31);
    int kt   = (int)((idx >> 9) & 7);
    int mt   = (int)((idx >> 12) & 3);
    int t    = (int)(idx >> 14);
    int b    = mt * 16 + (lane & 15);
    int i    = kt * 32 + ((q < 8) ? q : q + 8) + ((lane >= 16) ? 8 : 0);
    Xp[idx] = f2bf(x[((long)b * T_ + t) * I_ + i]);
}

// ---------------------------------------------------------------------------
// Device-wide ticket barrier (monotonic counter, memset to 0 each call).
// ---------------------------------------------------------------------------
__device__ __forceinline__ void grid_barrier(unsigned* cnt, unsigned nb) {
    __syncthreads();
    if (threadIdx.x == 0) {
        __threadfence();
        unsigned ticket = __hip_atomic_fetch_add(cnt, 1u, __ATOMIC_ACQ_REL,
                                                 __HIP_MEMORY_SCOPE_AGENT);
        unsigned target = (ticket / nb + 1u) * nb;
        while (__hip_atomic_load(cnt, __ATOMIC_ACQUIRE,
                                 __HIP_MEMORY_SCOPE_AGENT) < target) {
            __builtin_amdgcn_s_sleep(1);
        }
    }
    __syncthreads();
}

__device__ __forceinline__ float sigmoid_f(float x) {
    return 1.0f / (1.0f + __expf(-x));
}
__device__ __forceinline__ float tanh_f(float x) {
    float e = __expf(2.0f * x);
    return (e - 1.0f) / (e + 1.0f);
}

// ---------------------------------------------------------------------------
// One branch-free GEMM phase: nkt K-slices, A tiles from LDS, B tiles (4 gate
// columns) from L2-resident packed weights, 16 WMMAs per K-slice.
// ---------------------------------------------------------------------------
__device__ __forceinline__ void gemm_phase(const unsigned short* aBase, int aStride,
                                           const unsigned short* wBase, int wStride,
                                           int nkt, int lane, v8f (&acc)[4][4]) {
    for (int kt = 0; kt < nkt; ++kt) {
        v16bf a[4];
        #pragma unroll
        for (int mt = 0; mt < 4; ++mt)
            a[mt] = *(const v16bf*)(aBase + (mt * aStride + kt) * 512 + lane * 16);
        v16bf bt[4];
        #pragma unroll
        for (int g = 0; g < 4; ++g)
            bt[g] = *(const v16bf*)(wBase + (g * wStride + kt) * 512 + lane * 16);
        if (kt + 1 < nkt)   // warm the path for the next weight K-slice
            __builtin_prefetch(wBase + (kt + 1) * 512 + lane * 16, 0, 1);
        #pragma unroll
        for (int mt = 0; mt < 4; ++mt)
            #pragma unroll
            for (int g = 0; g < 4; ++g)
                acc[mt][g] = __builtin_amdgcn_wmma_f32_16x16x32_bf16(
                    false, a[mt], false, bt[g], (short)0, acc[mt][g], false, false);
    }
}

// ---------------------------------------------------------------------------
// Persistent 2-layer LSTM. Grid = 8 blocks x 256 threads (8 waves).
//   blocks 0..3 -> layer 0, blocks 4..7 -> layer 1;
//   wave w of block b owns hidden slab s = (b%4)*8 + w (16 cols, all 4 gates).
// Pipeline stage p: layer0 does t=p (p<512), layer1 does t=p-1 (p>=1); one
// grid barrier per stage. Cell state c stays in VGPRs for the whole run.
// Per stage the block's shared A image (x_t / h ping-pong, bf16, WMMA A
// layout) is staged into LDS once via async DMA and fed to WMMA by ds loads.
// ---------------------------------------------------------------------------
__global__ void __launch_bounds__(256)
lstm_persistent(const unsigned short* __restrict__ Xp,
                const unsigned short* __restrict__ W0p,
                const unsigned short* __restrict__ W1p,
                unsigned short* __restrict__ h0buf,   // 2 x 4*16*32*16 bf16
                unsigned short* __restrict__ h1buf,   // 2 x 4*16*32*16 bf16
                unsigned* __restrict__ barrier_cnt,
                const float* __restrict__ bx0,
                const float* __restrict__ bx1,
                const float* __restrict__ Wfc,
                const float* __restrict__ bfc,
                float* __restrict__ out) {
    extern __shared__ unsigned short smem[];   // 128 KB dynamic LDS

    const int tid   = threadIdx.x;
    const int lane  = tid & 31;
    const int wave  = tid >> 5;
    const int blk   = blockIdx.x;
    const int layer = blk >> 2;                  // 0 or 1
    const int s     = (blk & 3) * 8 + wave;      // slab 0..31
    const unsigned nblocks = gridDim.x;

    const int cloc  = lane & 15;                 // local column 0..15
    const int nh    = s * 16 + cloc;             // hidden column 0..511
    const int rowhi = (lane >> 4) & 1;           // lanes 16..31 hold rows M+8

    const float* bx = (layer == 0) ? bx0 : bx1;
    float bias[4];
    #pragma unroll
    for (int g = 0; g < 4; ++g) bias[g] = bx[g * H_ + nh];

    // Cell state: 4 M-tiles x 8 rows/lane, fp32, VGPR-resident for all 512 steps.
    v8f cst[4];
    #pragma unroll
    for (int mt = 0; mt < 4; ++mt)
        #pragma unroll
        for (int e = 0; e < 8; ++e) cst[mt][e] = 0.0f;

    const int KT = (layer == 0) ? 24 : 32;       // K/32 (768 or 1024)
    const unsigned short* wbase =
        ((layer == 0) ? W0p : W1p) + (long)s * 4 * KT * 512;

    // h-store geometry: hidden column nh becomes K of the consumer GEMM.
    const int kt_h = nh >> 5;
    const int k32  = nh & 31;
    const int hi   = (k32 >> 3) & 1;
    const int qst  = (k32 & 7) + ((k32 >> 4) << 3);

    // LDS region sizes (in shorts) for this layer's A image.
    const int n0 = (layer == 0) ? 16384 : 32768;   // x_t (32KB) or h0 (64KB)
    const int n1 = 32768;                          // h0 or h1 (64KB)

    for (int p = 0; p < T_ + 1; ++p) {
        const int  t      = (layer == 0) ? p : (p - 1);
        const bool active = (layer == 0) ? (p < T_) : (p >= 1);
        if (active) {
            // ---- Stage this block's A image into LDS (async DMA) ----------
            const unsigned short* h0rd =
                h0buf + (((layer == 0) ? ((t + 1) & 1) : (t & 1)) * 32768);
            const unsigned short* src0 = (layer == 0)
                ? (Xp + (long)t * (4 * 8 * 512)) : h0rd;
            const unsigned short* src1 = (layer == 0)
                ? h0rd : (h1buf + (((t + 1) & 1) * 32768));

            for (int i = tid * 8; i < n0; i += 256 * 8)
                copy16_g2l(src0 + i, smem + i);
            for (int i = tid * 8; i < n1; i += 256 * 8)
                copy16_g2l(src1 + i, smem + n0 + i);
            wait_g2l();
            __syncthreads();

            // ---- Gate GEMM: two branch-free phases ------------------------
            v8f acc[4][4];
            #pragma unroll
            for (int mt = 0; mt < 4; ++mt)
                #pragma unroll
                for (int g = 0; g < 4; ++g)
                    #pragma unroll
                    for (int e = 0; e < 8; ++e) acc[mt][g][e] = bias[g];

            if (layer == 0) {
                gemm_phase(smem,         8,  wbase,            24, 8,  lane, acc);
                gemm_phase(smem + 16384, 16, wbase + 8 * 512,  24, 16, lane, acc);
            } else {
                gemm_phase(smem,         16, wbase,            32, 16, lane, acc);
                gemm_phase(smem + 32768, 16, wbase + 16 * 512, 32, 16, lane, acc);
            }

            // ---- LSTM cell (wave-local: all 4 gates of this slab) ---------
            unsigned short* hwr =
                ((layer == 0) ? h0buf : h1buf) + ((t & 1) * 32768);
            const bool last = (layer == 1) && (t == T_ - 1);
            const float wf  = last ? Wfc[nh] : 0.0f;

            #pragma unroll
            for (int mt = 0; mt < 4; ++mt) {
                #pragma unroll
                for (int e = 0; e < 8; ++e) {
                    float ig = sigmoid_f(acc[mt][0][e]);
                    float fg = sigmoid_f(acc[mt][1][e]);
                    float og = sigmoid_f(acc[mt][2][e]);
                    float gg = tanh_f(acc[mt][3][e]);
                    float c  = fg * cst[mt][e] + ig * gg;
                    cst[mt][e] = c;
                    float h  = og * tanh_f(c);

                    int brow  = e + 8 * rowhi;            // row within M-tile
                    int lanep = brow + 16 * hi;           // dest lane in A tile
                    long sidx = (((long)mt * 16 + kt_h) * 32 + lanep) * 16 + qst;
                    hwr[sidx] = f2bf(h);

                    if (last) {                           // out = h1 @ Wfc + bfc
                        float v = h * wf;
                        if (s == 0 && cloc == 0) v += bfc[0];
                        atomicAdd(&out[16 * mt + brow], v);
                    }
                }
            }
        }
        grid_barrier(barrier_cnt, nblocks);
    }
}

// ---------------------------------------------------------------------------
// Host launcher. Workspace layout (bytes):
//   [0,256)            barrier counter
//   [256, +128K)       h0 ping-pong (2 x 64KB bf16, A-layout)
//   [.., +128K)        h1 ping-pong
//   [262400, +3MB)     W0 packed (768 x 2048 bf16, B-tile layout)
//   [.., +4MB)         W1 packed (1024 x 2048 bf16)
//   [.., +16MB)        X packed (T x 64 x 256 bf16, A-tile layout)
// ---------------------------------------------------------------------------
extern "C" void kernel_launch(void* const* d_in, const int* in_sizes, int n_in,
                              void* d_out, int out_size, void* d_ws, size_t ws_size,
                              hipStream_t stream) {
    (void)in_sizes; (void)n_in; (void)out_size; (void)ws_size;
    const float* x   = (const float*)d_in[0];
    const float* Wx0 = (const float*)d_in[1];
    const float* bx0 = (const float*)d_in[2];
    const float* Wh0 = (const float*)d_in[3];
    const float* Wx1 = (const float*)d_in[4];
    const float* bx1 = (const float*)d_in[5];
    const float* Wh1 = (const float*)d_in[6];
    const float* Wfc = (const float*)d_in[7];
    const float* bfc = (const float*)d_in[8];
    float* out = (float*)d_out;

    char* ws = (char*)d_ws;
    unsigned*       cnt   = (unsigned*)ws;
    unsigned short* h0buf = (unsigned short*)(ws + 256);
    unsigned short* h1buf = (unsigned short*)(ws + 256 + 131072);
    unsigned short* W0p   = (unsigned short*)(ws + 262400);
    unsigned short* W1p   = (unsigned short*)(ws + 262400 + 3145728L);
    unsigned short* Xp    = (unsigned short*)(ws + 262400 + 3145728L + 4194304L);

    (void)hipMemsetAsync(ws, 0, 262400, stream);
    (void)hipMemsetAsync(d_out, 0, B_ * sizeof(float), stream);

    {   // layer 0 weights: K = 256(x) + 512(h) = 768, KT = 24
        long total = 32L * 4 * 24 * 32 * 16;
        pack_w_kernel<<<(int)((total + 255) / 256), 256, 0, stream>>>(
            Wx0, Wh0, W0p, 256, 24);
    }
    {   // layer 1 weights: K = 512 + 512 = 1024, KT = 32
        long total = 32L * 4 * 32 * 32 * 16;
        pack_w_kernel<<<(int)((total + 255) / 256), 256, 0, stream>>>(
            Wx1, Wh1, W1p, 512, 32);
    }
    {   // packed input projections
        long total = (long)T_ * 4 * 8 * 32 * 16;
        pack_x_kernel<<<(int)((total + 255) / 256), 256, 0, stream>>>(x, Xp);
    }

    // 128 KB dynamic LDS per block (WGP has 320 KB; opt in above the default cap).
    (void)hipFuncSetAttribute((const void*)lstm_persistent,
                              hipFuncAttributeMaxDynamicSharedMemorySize, 131072);
    lstm_persistent<<<8, 256, 131072, stream>>>(Xp, W0p, W1p, h0buf, h1buf, cnt,
                                                bx0, bx1, Wfc, bfc, out);
}
